// Decoder_29678224015654
// MI455X (gfx1250) — matrix operand order repaired and checked by reading, not compiled, not run on hardware
//
#include <hip/hip_runtime.h>
#include <hip/hip_bf16.h>
#include <math.h>

// ---------------------------------------------------------------------------
// MI455X (gfx1250): bf16 WMMA GEMMs (v_wmma_f32_16x16x32_bf16, wave32) with
// fully padded (guard-free) tiles, native bf16 converts, fused LSTM cell
// kernels, log-softmax epilogue.
// ---------------------------------------------------------------------------

typedef __attribute__((ext_vector_type(16))) __bf16 v16bf;
typedef __attribute__((ext_vector_type(8)))  float  v8f;

#define TMT 128   // block tile M
#define TNT 128   // block tile N
#define TKC 32    // K chunk (one WMMA K per chunk)
#define TKP 40    // padded LDS row (80B = 20 banks: conflict-free, 16B aligned)

#define GF_ACC  1
#define GF_TANH 2

// native fp32 -> bf16 (backend emits hardware cvt on gfx1250, RNE)
__device__ __forceinline__ unsigned short f2bf(float x) {
  union { __bf16 h; unsigned short s; } cv;
  cv.h = (__bf16)x;
  return cv.s;
}

__device__ __forceinline__ float sigmoidf_(float x) {
  return 1.0f / (1.0f + __expf(-x));
}

union Frag16 { uint4 u[2]; v16bf v; };
union PackA  { __bf16 h[16]; uint4 u[2]; };

// C[crow0 ..][gridDim.x*128] (+)= A @ WT^T (+b1+b2, opt tanh)
// Preconditions (host-enforced): M%128==0, N%128==0, K%32==0, pointers
// 16B-aligned, lda%4==0 (floats), ldw%8==0 (halfs).  NO bounds checks.
__global__ __launch_bounds__(256) void gemm_bf16_wmma(
    const float* __restrict__ A, int lda,
    const unsigned short* __restrict__ WT, int ldw,
    const float* __restrict__ bias1, const float* __restrict__ bias2,
    float* __restrict__ C, int ldc, int crow0,
    int K, int flags)
{
  __shared__ alignas(16) unsigned short As[TMT][TKP];  // [m][k]
  __shared__ alignas(16) unsigned short Bs[TNT][TKP];  // [n][k]

  const int bm   = blockIdx.y * TMT;
  const int bn   = blockIdx.x * TNT;
  const int tid  = threadIdx.x;
  const int wave = tid >> 5;            // 8 waves
  const int lane = tid & 31;
  const int wm   = (wave & 3) * 32;     // 4 waves along M
  const int wn   = (wave >> 2) * 64;    // 2 waves along N
  const int lr   = lane & 15;
  const int lhi  = lane >> 4;

  v8f acc[2][4];
#pragma unroll
  for (int i = 0; i < 2; ++i)
#pragma unroll
    for (int j = 0; j < 4; ++j)
      acc[i][j] = (v8f){0.f,0.f,0.f,0.f,0.f,0.f,0.f,0.f};

  const int srow = tid >> 1;            // 0..127
  const int skb  = (tid & 1) * 16;      // 0 / 16

  const float*          Abase = A  + (size_t)(bm + srow) * lda + skb;
  const unsigned short* Bbase = WT + (size_t)(bn + srow) * ldw + skb;

  for (int k0 = 0; k0 < K; k0 += TKC) {
    __syncthreads();
    // ---- stage A tile: 16 fp32 -> 16 bf16 (native cvt), two b128 stores ----
    {
      const float4* Ap = reinterpret_cast<const float4*>(Abase + k0);
      PackA pa;
#pragma unroll
      for (int i = 0; i < 4; ++i) {
        const float4 a = Ap[i];
        pa.h[4*i + 0] = (__bf16)a.x;
        pa.h[4*i + 1] = (__bf16)a.y;
        pa.h[4*i + 2] = (__bf16)a.z;
        pa.h[4*i + 3] = (__bf16)a.w;
      }
      *reinterpret_cast<uint4*>(&As[srow][skb])     = pa.u[0];
      *reinterpret_cast<uint4*>(&As[srow][skb + 8]) = pa.u[1];
    }
    // ---- stage B tile: 16 bf16, straight b128 copies ----
    {
      const uint4* Bp = reinterpret_cast<const uint4*>(Bbase + k0);
      const uint4 b0 = Bp[0], b1 = Bp[1];
      *reinterpret_cast<uint4*>(&Bs[srow][skb])     = b0;
      *reinterpret_cast<uint4*>(&Bs[srow][skb + 8]) = b1;
      if (k0 + TKC < K) __builtin_prefetch(Bbase + k0 + TKC, 0, 1);
    }
    __syncthreads();

    // ---- WMMA fragments per ISA lane layout ----
    Frag16 af[2], bfr[4];
#pragma unroll
    for (int mt = 0; mt < 2; ++mt) {
      const int r  = wm + mt * 16 + lr;
      const int ak = lhi * 8;           // halves 0..7/8..15 then 16..23/24..31
      af[mt].u[0] = *reinterpret_cast<const uint4*>(&As[r][ak]);
      af[mt].u[1] = *reinterpret_cast<const uint4*>(&As[r][16 + ak]);
    }
#pragma unroll
    for (int nt = 0; nt < 4; ++nt) {
      const int n   = wn + nt * 16 + lr;
      const int kst = lhi * 8;          // halves 0..7/8..15 then 16..23/24..31
      bfr[nt].u[0] = *reinterpret_cast<const uint4*>(&Bs[n][kst]);
      bfr[nt].u[1] = *reinterpret_cast<const uint4*>(&Bs[n][kst + 16]);
    }
#pragma unroll
    for (int mt = 0; mt < 2; ++mt)
#pragma unroll
      for (int nt = 0; nt < 4; ++nt)
        acc[mt][nt] = __builtin_amdgcn_wmma_f32_16x16x32_bf16(
            false, af[mt].v, false, bfr[nt].v, (short)0, acc[mt][nt],
            false, false);
  }

  // ---- epilogue: element r -> row (lhi*8 + r), col (lane&15) ----
#pragma unroll
  for (int mt = 0; mt < 2; ++mt) {
#pragma unroll
    for (int nt = 0; nt < 4; ++nt) {
      const int gcol = bn + wn + nt * 16 + lr;
      float badd = 0.0f;
      if (bias1) badd += bias1[gcol];
      if (bias2) badd += bias2[gcol];
      float* cp = C + (size_t)(crow0 + bm + wm + mt * 16 + lhi * 8) * ldc + gcol;
#pragma unroll
      for (int r = 0; r < 8; ++r) {
        float v = acc[mt][nt][r] + badd;
        if (flags & GF_ACC)  v += cp[(size_t)r * ldc];
        if (flags & GF_TANH) v = tanhf(v);
        cp[(size_t)r * ldc] = v;
      }
    }
  }
}

// fp32 W rows [krow0, krow0+Kc) of [*, Nw] -> bf16 WT [Nout][Kpad], zero-padded
__global__ void k_transpose_bf16(const float* __restrict__ W, int Nw,
                                 int krow0, int Kc, int Kpad,
                                 unsigned short* __restrict__ WT, int Nout) {
  const int idx = blockIdx.x * blockDim.x + threadIdx.x;
  if (idx >= Nout * Kpad) return;
  const int n = idx / Kpad;
  const int k = idx - n * Kpad;
  float v = 0.0f;
  if (n < Nw && k < Kc) v = W[(size_t)(krow0 + k) * Nw + n];
  WT[idx] = f2bf(v);
}

// pad rows of [rows][Vin] fp32 to [rows][Vout] with zeros
__global__ void k_pad_rows(const float* __restrict__ src, float* __restrict__ dst,
                           int rows, int Vin, int Vout) {
  const int idx = blockIdx.x * blockDim.x + threadIdx.x;
  if (idx >= rows * Vout) return;
  const int r = idx / Vout;
  const int v = idx - r * Vout;
  dst[idx] = (v < Vin) ? src[(size_t)r * Vin + v] : 0.0f;
}

__global__ void k_set_sos(float* __restrict__ sos, int Bc, int Vout) {
  const int idx = blockIdx.x * blockDim.x + threadIdx.x;
  if (idx >= Bc * Vout) return;
  sos[idx] = ((idx % Vout) == 0) ? 1.0f : 0.0f;
}

// gate order i,f,g,o in G[rows][4H]; updates h,c in place; optional copy of h.
__global__ void k_lstm_cell(const float* __restrict__ G, float* __restrict__ h,
                            float* __restrict__ c, float* __restrict__ hcopy,
                            int rows, int H) {
  const int idx = blockIdx.x * blockDim.x + threadIdx.x;
  if (idx >= rows * H) return;
  const int r = idx / H;
  const int j = idx - r * H;
  const float* g = G + (size_t)r * 4 * H;
  const float iv = sigmoidf_(g[j]);
  const float fv = sigmoidf_(g[H + j]);
  const float gv = tanhf(g[2 * H + j]);
  const float ov = sigmoidf_(g[3 * H + j]);
  const float cc = fv * c[idx] + iv * gv;
  const float hh = ov * tanhf(cc);
  c[idx] = cc;
  h[idx] = hh;
  if (hcopy) hcopy[idx] = hh;
}

// logits rows ordered (s, u, b), row stride ldl; fc bias folded in here.
__global__ __launch_bounds__(256) void k_log_softmax_out(
    const float* __restrict__ logits, int ldl, const float* __restrict__ bias,
    float* __restrict__ out, int Vc, int Bc, int SUBc, int SEQc) {
  const int row = blockIdx.x;
  const float* L = logits + (size_t)row * ldl;
  __shared__ float sd[256];
  const int tid = threadIdx.x;

  float m = -3.4e38f;
  for (int v = tid; v < Vc; v += 256) m = fmaxf(m, L[v] + bias[v]);
  sd[tid] = m; __syncthreads();
  for (int st = 128; st > 0; st >>= 1) {
    if (tid < st) sd[tid] = fmaxf(sd[tid], sd[tid + st]);
    __syncthreads();
  }
  const float mx = sd[0];
  __syncthreads();

  float sum = 0.0f;
  for (int v = tid; v < Vc; v += 256) sum += __expf(L[v] + bias[v] - mx);
  sd[tid] = sum; __syncthreads();
  for (int st = 128; st > 0; st >>= 1) {
    if (tid < st) sd[tid] += sd[tid + st];
    __syncthreads();
  }
  const float ls = __logf(sd[0]) + mx;

  const int sstep = row / (2 * Bc);
  const int rr    = row - sstep * 2 * Bc;
  const int u     = rr / Bc;
  const int b     = rr - u * Bc;
  const int t     = u * SUBc + sstep;
  float* op = out + (size_t)b * SEQc * Vc + (size_t)t * Vc;
  for (int v = tid; v < Vc; v += 256) op[v] = L[v] + bias[v] - ls;
}

extern "C" void kernel_launch(void* const* d_in, const int* in_sizes, int n_in,
                              void* d_out, int out_size, void* d_ws, size_t ws_size,
                              hipStream_t stream) {
  (void)in_sizes; (void)n_in; (void)out_size; (void)ws_size;
  const float* z    = (const float*)d_in[0];
  const float* x    = (const float*)d_in[1];
  const float* cin  = (const float*)d_in[2];   // [1,B,V] -> [512,258]
  const float* ci_W = (const float*)d_in[3];
  const float* ci_b = (const float*)d_in[4];
  const float* cW1i = (const float*)d_in[5];
  const float* cW1h = (const float*)d_in[6];
  const float* cb1i = (const float*)d_in[7];
  const float* cb1h = (const float*)d_in[8];
  const float* cW2i = (const float*)d_in[9];
  const float* cW2h = (const float*)d_in[10];
  const float* cb2i = (const float*)d_in[11];
  const float* cb2h = (const float*)d_in[12];
  const float* co_W = (const float*)d_in[13];
  const float* co_b = (const float*)d_in[14];
  const float* di_W = (const float*)d_in[15];
  const float* di_b = (const float*)d_in[16];
  const float* dW1i = (const float*)d_in[17];
  const float* dW1h = (const float*)d_in[18];
  const float* db1i = (const float*)d_in[19];
  const float* db1h = (const float*)d_in[20];
  const float* dW2i = (const float*)d_in[21];
  const float* dW2h = (const float*)d_in[22];
  const float* db2i = (const float*)d_in[23];
  const float* db2h = (const float*)d_in[24];
  const float* fc_W = (const float*)d_in[25];
  const float* fc_b = (const float*)d_in[26];
  float* out = (float*)d_out;

  const int Bc = 512, Vc = 258, Vp = 288, CHc = 1024, COc = 512, DHc = 1024;
  const int SUBc = 16, SEQc = 32, Zc = 512, Np = 384;  // fc N padded to 384

  // ---- workspace bump allocator ----
  char* base = (char*)d_ws;
  size_t off = 0;
  auto alloc = [&](size_t bytes) -> void* {
    void* p = base + off;
    off = (off + bytes + 255) & ~(size_t)255;
    return p;
  };
  // bf16 transposed weights [N][Kpad]
  unsigned short* ciT    = (unsigned short*)alloc((size_t)CHc   * Zc  * 2);
  unsigned short* cW1iT  = (unsigned short*)alloc((size_t)4*CHc * Vp  * 2);
  unsigned short* cW1hT  = (unsigned short*)alloc((size_t)4*CHc * CHc * 2);
  unsigned short* cW2iT  = (unsigned short*)alloc((size_t)4*CHc * CHc * 2);
  unsigned short* cW2hT  = (unsigned short*)alloc((size_t)4*CHc * CHc * 2);
  unsigned short* coT    = (unsigned short*)alloc((size_t)COc   * CHc * 2);
  unsigned short* diT    = (unsigned short*)alloc((size_t)DHc   * COc * 2);
  unsigned short* dW1iTa = (unsigned short*)alloc((size_t)4*DHc * COc * 2);
  unsigned short* dW1iTb = (unsigned short*)alloc((size_t)4*DHc * Vp  * 2);
  unsigned short* dW1hT  = (unsigned short*)alloc((size_t)4*DHc * DHc * 2);
  unsigned short* dW2iT  = (unsigned short*)alloc((size_t)4*DHc * DHc * 2);
  unsigned short* dW2hT  = (unsigned short*)alloc((size_t)4*DHc * DHc * 2);
  unsigned short* fcT    = (unsigned short*)alloc((size_t)Np    * DHc * 2);
  // fp32 activations
  float* hc1    = (float*)alloc((size_t)Bc*CHc*4);
  float* cc1    = (float*)alloc((size_t)Bc*CHc*4);
  float* hc2    = (float*)alloc((size_t)Bc*CHc*4);
  float* cc2    = (float*)alloc((size_t)Bc*CHc*4);
  float* Xic    = (float*)alloc((size_t)Bc*4*CHc*4);
  float* chb    = (float*)alloc((size_t)2*Bc*CHc*4);      // [t][b][CH]
  float* cbuf   = (float*)alloc((size_t)2*Bc*COc*4);      // [u][b][CO]
  float* hd1    = (float*)alloc((size_t)2*Bc*DHc*4);
  float* cd1    = (float*)alloc((size_t)2*Bc*DHc*4);
  float* hd2    = (float*)alloc((size_t)2*Bc*DHc*4);
  float* cd2    = (float*)alloc((size_t)2*Bc*DHc*4);
  float* Xc     = (float*)alloc((size_t)2*Bc*4*DHc*4);
  float* G      = (float*)alloc((size_t)2*Bc*4*DHc*4);
  float* xpad   = (float*)alloc((size_t)SEQc*Bc*Vp*4);    // x padded 258->288
  float* cinPad = (float*)alloc((size_t)Bc*Vp*4);
  float* sosPad = (float*)alloc((size_t)Bc*Vp*4);
  float* hs     = (float*)alloc((size_t)SUBc*2*Bc*DHc*4); // [s][u][b][DH]
  float* lg     = (float*)alloc((size_t)SUBc*2*Bc*Np*4);  // row stride 384

  auto gemm = [&](const float* A, int lda, const unsigned short* WT, int ldw,
                  const float* b1, const float* b2, float* C, int ldc,
                  int crow0, int M, int N, int K, int flags) {
    dim3 grid(N / TNT, M / TMT);
    gemm_bf16_wmma<<<grid, 256, 0, stream>>>(A, lda, WT, ldw, b1, b2,
                                             C, ldc, crow0, K, flags);
  };
  auto cell = [&](float* Gp, float* h, float* c, float* hcopy, int rows, int H) {
    int total = rows * H;
    k_lstm_cell<<<(total + 255) / 256, 256, 0, stream>>>(Gp, h, c, hcopy, rows, H);
  };
  auto cvt = [&](const float* W, int Nw, int krow0, int Kc, int Kpad,
                 unsigned short* WT, int Nout) {
    int total = Nout * Kpad;
    k_transpose_bf16<<<(total + 255) / 256, 256, 0, stream>>>(W, Nw, krow0, Kc,
                                                              Kpad, WT, Nout);
  };

  // ---- one-time per launch: weight convert+transpose to bf16 (zero-padded K) ----
  cvt(ci_W, CHc,   0,   Zc,  Zc,  ciT,    CHc);
  cvt(cW1i, 4*CHc, 0,   Vc,  Vp,  cW1iT,  4*CHc);
  cvt(cW1h, 4*CHc, 0,   CHc, CHc, cW1hT,  4*CHc);
  cvt(cW2i, 4*CHc, 0,   CHc, CHc, cW2iT,  4*CHc);
  cvt(cW2h, 4*CHc, 0,   CHc, CHc, cW2hT,  4*CHc);
  cvt(co_W, COc,   0,   CHc, CHc, coT,    COc);
  cvt(di_W, DHc,   0,   COc, COc, diT,    DHc);
  cvt(dW1i, 4*DHc, 0,   COc, COc, dW1iTa, 4*DHc);
  cvt(dW1i, 4*DHc, COc, Vc,  Vp,  dW1iTb, 4*DHc);
  cvt(dW1h, 4*DHc, 0,   DHc, DHc, dW1hT,  4*DHc);
  cvt(dW2i, 4*DHc, 0,   DHc, DHc, dW2iT,  4*DHc);
  cvt(dW2h, 4*DHc, 0,   DHc, DHc, dW2hT,  4*DHc);
  cvt(fc_W, Vc,    0,   DHc, DHc, fcT,    Np);   // rows 258..383 zeroed

  k_pad_rows<<<(SEQc*Bc*Vp + 255) / 256, 256, 0, stream>>>(x, xpad, SEQc*Bc, Vc, Vp);
  k_pad_rows<<<(Bc*Vp + 255) / 256, 256, 0, stream>>>(cin, cinPad, Bc, Vc, Vp);
  k_set_sos<<<(Bc*Vp + 255) / 256, 256, 0, stream>>>(sosPad, Bc, Vp);

  // ---- conductor ----
  gemm(z, Zc, ciT, Zc, ci_b, nullptr, hc1, CHc, 0, Bc, CHc, Zc, GF_TANH);
  hipMemsetAsync(cc1, 0, (size_t)Bc*CHc*4, stream);
  hipMemsetAsync(hc2, 0, (size_t)Bc*CHc*4, stream);
  hipMemsetAsync(cc2, 0, (size_t)Bc*CHc*4, stream);
  // step-invariant layer-1 input GEMM (+ both biases)
  gemm(cinPad, Vp, cW1iT, Vp, cb1i, cb1h, Xic, 4*CHc, 0, Bc, 4*CHc, Vp, 0);
  for (int t = 0; t < 2; ++t) {
    hipMemcpyAsync(G, Xic, (size_t)Bc*4*CHc*4, hipMemcpyDeviceToDevice, stream);
    gemm(hc1, CHc, cW1hT, CHc, nullptr, nullptr, G, 4*CHc, 0, Bc, 4*CHc, CHc, GF_ACC);
    cell(G, hc1, cc1, nullptr, Bc, CHc);
    gemm(hc1, CHc, cW2iT, CHc, cb2i, cb2h, G, 4*CHc, 0, Bc, 4*CHc, CHc, 0);
    gemm(hc2, CHc, cW2hT, CHc, nullptr, nullptr, G, 4*CHc, 0, Bc, 4*CHc, CHc, GF_ACC);
    cell(G, hc2, cc2, chb + (size_t)t*Bc*CHc, Bc, CHc);
  }
  // c = ch @ co_W + co_b   (rows = U*B)
  gemm(chb, CHc, coT, CHc, co_b, nullptr, cbuf, COc, 0, 2*Bc, COc, CHc, 0);

  // ---- decoder init (both segments batched, rows = 2*B) ----
  gemm(cbuf, COc, diT, COc, di_b, nullptr, hd1, DHc, 0, 2*Bc, DHc, COc, GF_TANH);
  hipMemsetAsync(cd1, 0, (size_t)2*Bc*DHc*4, stream);
  hipMemsetAsync(hd2, 0, (size_t)2*Bc*DHc*4, stream);
  hipMemsetAsync(cd2, 0, (size_t)2*Bc*DHc*4, stream);
  // step-invariant part of layer-1 input: c @ dW1i[0:CO] + biases
  gemm(cbuf, COc, dW1iTa, COc, db1i, db1h, Xc, 4*DHc, 0, 2*Bc, 4*DHc, COc, 0);

  // ---- 16 sequential decoder steps (segments u=0,1 run as one batch) ----
  for (int s = 0; s < SUBc; ++s) {
    hipMemcpyAsync(G, Xc, (size_t)2*Bc*4*DHc*4, hipMemcpyDeviceToDevice, stream);
    const float* prev0 = (s == 0) ? sosPad : (xpad + (size_t)(s - 1) * Bc * Vp);
    const float* prev1 = xpad + (size_t)(SUBc - 1 + s) * Bc * Vp;
    gemm(prev0, Vp, dW1iTb, Vp, nullptr, nullptr, G, 4*DHc, 0,  Bc, 4*DHc, Vp, GF_ACC);
    gemm(prev1, Vp, dW1iTb, Vp, nullptr, nullptr, G, 4*DHc, Bc, Bc, 4*DHc, Vp, GF_ACC);
    gemm(hd1, DHc, dW1hT, DHc, nullptr, nullptr, G, 4*DHc, 0, 2*Bc, 4*DHc, DHc, GF_ACC);
    cell(G, hd1, cd1, nullptr, 2*Bc, DHc);
    gemm(hd1, DHc, dW2iT, DHc, db2i, db2h, G, 4*DHc, 0, 2*Bc, 4*DHc, DHc, 0);
    gemm(hd2, DHc, dW2hT, DHc, nullptr, nullptr, G, 4*DHc, 0, 2*Bc, 4*DHc, DHc, GF_ACC);
    cell(G, hd2, cd2, hs + (size_t)s*2*Bc*DHc, 2*Bc, DHc);
  }

  // ---- final projection (N padded to 384, bias folded into softmax) ----
  gemm(hs, DHc, fcT, DHc, nullptr, nullptr, lg, Np, 0, SUBc*2*Bc, Np, DHc, 0);
  k_log_softmax_out<<<SUBc*2*Bc, 256, 0, stream>>>(lg, Np, fc_b, out,
                                                   Vc, Bc, SUBc, SEQc);
}
